// MultiHeadAttention_S_88046829568026
// MI455X (gfx1250) — matrix operand order repaired and checked by reading, not compiled, not run on hardware
//
#include <hip/hip_runtime.h>
#include <hip/hip_bf16.h>
#include <stdint.h>

// ---------------------------------------------------------------------------
// MultiHeadAttention-over-spatial (B=8,S=12,C=128,H=W=32, HEADS=8, QK=64,DH=8)
// CDNA5 (gfx1250): bf16 WMMA 16x16x32 for all GEMMs, weight fragments resident
// in VGPRs, per-sample activations resident in the 320KB WGP LDS, bf16
// intermediates kept L2-resident (192MB), async global->LDS staging via
// GLOBAL_LOAD_ASYNC_TO_LDS_B128 (ASYNCcnt).
// ---------------------------------------------------------------------------

#define Bv     8
#define Sv     12
#define Cv     128
#define HWv    1024
#define QKv    64
#define HEADSv 8
#define DHv    8
#define Dv     (DHv * HWv)      // 8192
#define BSv    (Bv * Sv)        // 96
#define BHv    (Bv * HEADSv)    // 64

#if __has_builtin(__builtin_amdgcn_global_load_async_to_lds_b128) && \
    __has_builtin(__builtin_amdgcn_s_wait_asynccnt)
#define USE_ASYNC_COPY 1
#else
#define USE_ASYNC_COPY 0
#endif

typedef __attribute__((ext_vector_type(16))) __bf16 v16bf;
typedef __attribute__((ext_vector_type(8)))  float  v8f;
typedef int v4i32 __attribute__((vector_size(16)));   // matches builtin param

struct Frag { union { v16bf v; uint32_t u[8]; }; };

__device__ __forceinline__ unsigned short f2bf(float f) {
    uint32_t u = __float_as_uint(f);
    u += 0x7FFFu + ((u >> 16) & 1u);      // round-to-nearest-even
    return (unsigned short)(u >> 16);
}
__device__ __forceinline__ float bf2f(unsigned short h) {
    return __uint_as_float(((uint32_t)h) << 16);
}

#if USE_ASYNC_COPY
// gfx1250 async copy: 16B per lane, global -> LDS, tracked by ASYNCcnt.
// Param types per compiler diagnostic: (v4i32 AS1*, v4i32 AS3*, imm, imm).
__device__ __forceinline__ void async_cp16(void* lds, const void* g) {
    __builtin_amdgcn_global_load_async_to_lds_b128(
        (__attribute__((address_space(1))) v4i32*)(uintptr_t)g,
        (__attribute__((address_space(3))) v4i32*)(uint32_t)(uintptr_t)lds, 0, 0);
}
#endif

// A-matrix 16x32 bf16, storage row-major [m][K] (contiguous k order, same as load_frag_B):
// lane m (0-15): K = {0..15}; lane m+16: K = {16..31}, pairs/VGPR.
__device__ __forceinline__ void load_frag_A(Frag& f, const unsigned short* base,
                                            int row_stride, int lane) {
    const int half = (lane >> 4) & 1, r = lane & 15;
    const unsigned short* p = base + r * row_stride;
#pragma unroll
    for (int i = 0; i < 8; ++i) {
        int k = 2 * i + 16 * half;
        f.u[i] = *(const uint32_t*)(p + k);
    }
}
// B-matrix 32x16 bf16 from N-major storage [n][K]: lane n, VGPR i holds
// (K=2i,2i+1) for lanes 0-15 and (16+2i,17+2i) for lanes 16-31.
__device__ __forceinline__ void load_frag_B(Frag& f, const unsigned short* base,
                                            int row_stride, int lane) {
    const int half = (lane >> 4) & 1, n = lane & 15;
    const unsigned short* p = base + n * row_stride;
#pragma unroll
    for (int i = 0; i < 8; ++i) {
        int k = 2 * i + 16 * half;
        f.u[i] = *(const uint32_t*)(p + k);
    }
}
// B-matrix 32x16 bf16 from K-major storage [k][n] (pack pairs manually).
__device__ __forceinline__ void load_frag_B_kmajor(Frag& f, const unsigned short* base,
                                                   int row_stride, int lane) {
    const int half = (lane >> 4) & 1, n = lane & 15;
#pragma unroll
    for (int i = 0; i < 8; ++i) {
        int k = 2 * i + 16 * half;
        uint32_t lo = base[k * row_stride + n];
        uint32_t hi = base[(k + 1) * row_stride + n];
        f.u[i] = lo | (hi << 16);
    }
}

__device__ __forceinline__ v8f wmma_bf16(const Frag& a, const Frag& b, v8f c) {
    return __builtin_amdgcn_wmma_f32_16x16x32_bf16(false, a.v, false, b.v,
                                                   (short)0, c, false, false);
}

// ---------------------------------------------------------------------------
// Kernel 1: fused conv1x1 (W[64,128] @ x[128,1024]) + GroupNorm(1) for q/k/v.
// Weight fragments (4 mtiles x 4 ksteps) hoisted into VGPRs; y kept fully in
// LDS (128 KB bf16); normalized result scattered to head-major layout (bf16).
// grid = (96, 3), block = 256 (8 wave32), dynamic LDS = 184832 B.
// ---------------------------------------------------------------------------
#define XSTR (Cv + 8)     // 136, padded stride for transposed x staging

__global__ __launch_bounds__(256) void proj_kernel(
    const float* __restrict__ qx, const float* __restrict__ kx, const float* __restrict__ vx,
    const float* __restrict__ Wq, const float* __restrict__ gq, const float* __restrict__ bq,
    const float* __restrict__ Wk, const float* __restrict__ gk, const float* __restrict__ bk,
    const float* __restrict__ Wv, const float* __restrict__ gv, const float* __restrict__ bv,
    unsigned short* __restrict__ qh, unsigned short* __restrict__ kh,
    unsigned short* __restrict__ vh)
{
    extern __shared__ unsigned char smem[];
    unsigned short* Wb  = (unsigned short*)smem;          // [64][128]     16384 B
    unsigned short* xs  = Wb + QKv * Cv;                  // [128][136]    34816 B
    unsigned short* yb  = xs + 128 * XSTR;                // [64][1024]   131072 B
    float*          red = (float*)(yb + QKv * HWv);       // [512]          2048 B
    float*          gms = red + 512;                      // [64]
    float*          bts = gms + QKv;                      // [64]

    const int bs = blockIdx.x;          // sample 0..95
    const int pj = blockIdx.y;          // 0=q 1=k 2=v
    const float *x, *Wp, *gm, *bt; unsigned short* outp;
    if      (pj == 0) { x = qx; Wp = Wq; gm = gq; bt = bq; outp = qh; }
    else if (pj == 1) { x = kx; Wp = Wk; gm = gk; bt = bk; outp = kh; }
    else              { x = vx; Wp = Wv; gm = gv; bt = bv; outp = vh; }
    x += (size_t)bs * Cv * HWv;

    const int tid = threadIdx.x, lane = tid & 31, wave = tid >> 5;

    for (int i = tid; i < QKv * Cv; i += 256) Wb[i] = f2bf(Wp[i]);
    if (tid < QKv) { gms[tid] = gm[tid]; bts[tid] = bt[tid]; }
    __syncthreads();

    // Hoist weight fragments for all (mtile, kstep) -> 128 VGPRs, loaded once.
    Frag aw[4][4];
#pragma unroll
    for (int mt = 0; mt < 4; ++mt)
#pragma unroll
        for (int kb = 0; kb < 4; ++kb)
            load_frag_A(aw[mt][kb], Wb + (mt * 16) * Cv + kb * 32, Cv, lane);

    const int nt = wave;                // each wave owns one 16-pixel column tile
    float sum = 0.f, ssq = 0.f;
    for (int n0 = 0; n0 < HWv; n0 += 128) {
        // stage 128 channels x 128 pixels, transposed to [pixel][channel] bf16
        for (int i = tid; i < Cv * 128; i += 256) {
            int c = i >> 7, p = i & 127;                 // coalesced over p
            xs[p * XSTR + c] = f2bf(x[c * HWv + n0 + p]);
        }
        if (n0 + 128 < HWv) {                            // gfx1250 global_prefetch_b8
            int c = tid >> 1, p = (tid & 1) * 64;
            __builtin_prefetch(&x[c * HWv + n0 + 128 + p], 0, 1);
        }
        __syncthreads();

        Frag bx[4];
#pragma unroll
        for (int kb = 0; kb < 4; ++kb)
            load_frag_B(bx[kb], xs + (nt * 16) * XSTR + kb * 32, XSTR, lane);
#pragma unroll
        for (int mt = 0; mt < 4; ++mt) {
            v8f acc = {};
#pragma unroll
            for (int kb = 0; kb < 4; ++kb) acc = wmma_bf16(aw[mt][kb], bx[kb], acc);
            int n = n0 + nt * 16 + (lane & 15);
            int mbase = mt * 16 + ((lane >> 4) & 1) * 8;
#pragma unroll
            for (int r = 0; r < 8; ++r) {
                float val = acc[r];
                sum += val; ssq += val * val;
                yb[(mbase + r) * HWv + n] = f2bf(val);
            }
        }
        __syncthreads();
    }

    // GroupNorm(1) stats over all 65536 values
    red[tid] = sum; red[256 + tid] = ssq;
    __syncthreads();
    for (int s = 128; s > 0; s >>= 1) {
        if (tid < s) { red[tid] += red[tid + s]; red[256 + tid] += red[256 + tid + s]; }
        __syncthreads();
    }
    const float mean = red[0] * (1.0f / 65536.0f);
    const float var  = red[256] * (1.0f / 65536.0f) - mean * mean;
    const float rstd = rsqrtf(var + 1e-5f);

    // normalize + affine, scatter to [b*H+h][s][dh*1024+n] bf16 (stays in L2)
    const int b = bs / Sv, s_ = bs % Sv;
    for (int i = tid; i < QKv * HWv; i += 256) {
        int m = i >> 10, n = i & 1023;
        float o = (bf2f(yb[i]) - mean) * rstd * gms[m] + bts[m];
        int h = m >> 3, dh = m & 7;
        size_t dst = (((size_t)(b * HEADSv + h) * Sv) + s_) * Dv + (size_t)dh * HWv + n;
        outp[dst] = f2bf(o);
    }
}

// ---------------------------------------------------------------------------
// Kernel 2: attention per (b,head). S padded 12->16. Scores via wmma with the
// K=8192 reduction split across 8 waves (LDS reduce), softmax fp32, then
// O = P @ V via wmma (P zero-padded 16x32). Staging uses gfx1250 async
// global->LDS B128 copies when available.  grid = 64, block = 256.
// ---------------------------------------------------------------------------
#define CSTR 264     // 256-wide chunk + pad (row stride 528 B = 33*16, 16B ok)

__global__ __launch_bounds__(256) void attn_kernel(
    const unsigned short* __restrict__ qh, const unsigned short* __restrict__ kh,
    const unsigned short* __restrict__ vh, unsigned short* __restrict__ o_ws)
{
    __shared__ __align__(16) unsigned short qs[16][CSTR];
    __shared__ __align__(16) unsigned short ks[16][CSTR];
    __shared__ __align__(16) unsigned short vs[32][CSTR];
    __shared__ float          sred[8][256];
    __shared__ float          smat[16][16];
    __shared__ unsigned short ps[16][32];

    const int bh = blockIdx.x, tid = threadIdx.x, lane = tid & 31, wave = tid >> 5;
    const unsigned short* qb = qh + (size_t)bh * Sv * Dv;
    const unsigned short* kb = kh + (size_t)bh * Sv * Dv;
    const unsigned short* vb = vh + (size_t)bh * Sv * Dv;

    // zero V pad rows (k = 12..31) once: multiplied by zero P cols, avoid 0*NaN
    for (int i = tid; i < 20 * CSTR; i += 256) vs[12 + i / CSTR][i % CSTR] = 0;

    // ---- scores = q @ k^T, K split: wave w covers k0 + [w*32, w*32+32) ----
    v8f sacc = {};
    for (int k0 = 0; k0 < Dv; k0 += 256) {
#if USE_ASYNC_COPY
        for (int i = tid; i < Sv * 32; i += 256) {       // 12 rows x 32 x 8 elems
            int r = i >> 5, c8 = (i & 31) << 3;
            async_cp16(&qs[r][c8], qb + r * Dv + k0 + c8);
            async_cp16(&ks[r][c8], kb + r * Dv + k0 + c8);
        }
        __builtin_amdgcn_s_wait_asynccnt(0);
#else
        for (int i = tid; i < Sv * 256; i += 256) {
            int r = i >> 8, c = i & 255;
            qs[r][c] = qb[r * Dv + k0 + c];
            ks[r][c] = kb[r * Dv + k0 + c];
        }
#endif
        __syncthreads();
        Frag a, b;
        load_frag_A(a, &qs[0][0] + wave * 32, CSTR, lane);
        load_frag_B(b, &ks[0][0] + wave * 32, CSTR, lane);
        sacc = wmma_bf16(a, b, sacc);
        __syncthreads();
    }
#pragma unroll
    for (int r = 0; r < 8; ++r) sred[wave][r * 32 + lane] = sacc[r];
    __syncthreads();
    {   // cross-wave reduce -> smat (scaled)
        float sv = 0.f;
        for (int w = 0; w < 8; ++w) sv += sred[w][tid];
        int r = tid >> 5, l = tid & 31;
        int m = r + 8 * (l >> 4), n = l & 15;
        smat[m][n] = sv * 0.0110485435f;     // 1/sqrt(8192)
    }
    __syncthreads();
    for (int i = tid; i < 16 * 32; i += 256) ps[i >> 5][i & 31] = 0;
    __syncthreads();
    if (tid < Sv) {                           // fp32 softmax over 12 keys
        float mx = -1e30f;
        for (int t = 0; t < Sv; ++t) mx = fmaxf(mx, smat[tid][t]);
        float den = 0.f, e[Sv];
        for (int t = 0; t < Sv; ++t) { e[t] = __expf(smat[tid][t] - mx); den += e[t]; }
        float inv = 1.0f / den;
        for (int t = 0; t < Sv; ++t) ps[tid][t] = f2bf(e[t] * inv);
    }
    __syncthreads();

    Frag pa;
    load_frag_A(pa, &ps[0][0], 32, lane);     // P as 16x32 A-operand (zero padded)

    // ---- O = P @ V, stream N = 8192 in chunks of 256 pixels ----
    const int b = bh >> 3, h = bh & 7;
    for (int n0 = 0; n0 < Dv; n0 += 256) {
#if USE_ASYNC_COPY
        for (int i = tid; i < Sv * 32; i += 256) {
            int r = i >> 5, c8 = (i & 31) << 3;
            async_cp16(&vs[r][c8], vb + r * Dv + n0 + c8);
        }
        __builtin_amdgcn_s_wait_asynccnt(0);
#else
        for (int i = tid; i < Sv * 256; i += 256) {
            int r = i >> 8, c = i & 255;
            vs[r][c] = vb[r * Dv + n0 + c];
        }
#endif
        __syncthreads();
        for (int t = 0; t < 2; ++t) {
            int nt = wave * 2 + t;
            Frag bf2;
            load_frag_B_kmajor(bf2, &vs[0][0] + nt * 16, CSTR, lane);
            v8f acc = {};
            acc = wmma_bf16(pa, bf2, acc);
            int d0 = n0 + nt * 16;
            int dh = d0 >> 10, pix = (d0 & 1023) + (lane & 15);
            int mbase = 8 * ((lane >> 4) & 1);
#pragma unroll
            for (int r = 0; r < 8; ++r) {
                int m = mbase + r;                        // query position s
                if (m < Sv) {
                    size_t dst = (((size_t)(b * Sv + m) * QKv) + h * DHv + dh) * HWv + pix;
                    o_ws[dst] = f2bf(acc[r]);
                }
            }
        }
        __syncthreads();
    }
}

// ---------------------------------------------------------------------------
// Kernel 3: fused conv1x1 (Wo[128,64] @ o[64,1024]) + GroupNorm(1,128) + SiLU.
// Weight fragments (8 mtiles x 2 ksteps) hoisted into VGPRs; whole 128x1024
// result resident in LDS (256 KB bf16). grid = 96, block = 256,
// dynamic LDS = 300032 B.
// ---------------------------------------------------------------------------
#define OSTR 72      // 64 + 8 pad

__global__ __launch_bounds__(256) void outproj_kernel(
    const unsigned short* __restrict__ o_ws, const float* __restrict__ Wo,
    const float* __restrict__ go, const float* __restrict__ bo,
    float* __restrict__ out)
{
    extern __shared__ unsigned char smem[];
    unsigned short* Wb  = (unsigned short*)smem;          // [128][64]    16384 B
    unsigned short* os  = Wb + Cv * QKv;                  // [128][72]    18432 B
    unsigned short* yb  = os + 128 * OSTR;                // [128][1024] 262144 B
    float*          red = (float*)(yb + Cv * HWv);        // [512]         2048 B
    float*          gms = red + 512;                      // [128]
    float*          bts = gms + Cv;                       // [128]

    const int bs = blockIdx.x, tid = threadIdx.x, lane = tid & 31, wave = tid >> 5;
    const unsigned short* ob = o_ws + (size_t)bs * QKv * HWv;

    for (int i = tid; i < Cv * QKv; i += 256) Wb[i] = f2bf(Wo[i]);
    if (tid < Cv) { gms[tid] = go[tid]; bts[tid] = bo[tid]; }
    __syncthreads();

    Frag aw[8][2];
#pragma unroll
    for (int mt = 0; mt < 8; ++mt)
#pragma unroll
        for (int kb = 0; kb < 2; ++kb)
            load_frag_A(aw[mt][kb], Wb + (mt * 16) * QKv + kb * 32, QKv, lane);

    const int nt = wave;
    float sum = 0.f, ssq = 0.f;
    for (int n0 = 0; n0 < HWv; n0 += 128) {
        for (int i = tid; i < QKv * 128; i += 256) {      // transpose-stage bf16
            int qc = i >> 7, p = i & 127;
            os[p * OSTR + qc] = ob[qc * HWv + n0 + p];
        }
        __syncthreads();

        Frag bx[2];
#pragma unroll
        for (int kb = 0; kb < 2; ++kb)
            load_frag_B(bx[kb], os + (nt * 16) * OSTR + kb * 32, OSTR, lane);
#pragma unroll
        for (int mt = 0; mt < 8; ++mt) {
            v8f acc = {};
#pragma unroll
            for (int kb = 0; kb < 2; ++kb) acc = wmma_bf16(aw[mt][kb], bx[kb], acc);
            int n = n0 + nt * 16 + (lane & 15);
            int mbase = mt * 16 + ((lane >> 4) & 1) * 8;
#pragma unroll
            for (int r = 0; r < 8; ++r) {
                float val = acc[r];
                sum += val; ssq += val * val;
                yb[(mbase + r) * HWv + n] = f2bf(val);
            }
        }
        __syncthreads();
    }

    red[tid] = sum; red[256 + tid] = ssq;
    __syncthreads();
    for (int s = 128; s > 0; s >>= 1) {
        if (tid < s) { red[tid] += red[tid + s]; red[256 + tid] += red[256 + tid + s]; }
        __syncthreads();
    }
    const float mean = red[0] * (1.0f / 131072.0f);
    const float var  = red[256] * (1.0f / 131072.0f) - mean * mean;
    const float rstd = rsqrtf(var + 1e-5f);

    float* outb = out + (size_t)bs * Cv * HWv;
    for (int i = tid; i < Cv * HWv; i += 256) {
        int c = i >> 10;
        float y = (bf2f(yb[i]) - mean) * rstd * gms[c] + bts[c];
        outb[i] = y / (1.0f + __expf(-y));                // SiLU
    }
}

// ---------------------------------------------------------------------------
extern "C" void kernel_launch(void* const* d_in, const int* in_sizes, int n_in,
                              void* d_out, int out_size, void* d_ws, size_t ws_size,
                              hipStream_t stream) {
    (void)in_sizes; (void)n_in; (void)out_size; (void)ws_size;
    const float* q  = (const float*)d_in[0];
    const float* k  = (const float*)d_in[1];
    const float* v  = (const float*)d_in[2];
    const float* Wq = (const float*)d_in[3];
    const float* gq = (const float*)d_in[4];
    const float* bq = (const float*)d_in[5];
    const float* Wk = (const float*)d_in[6];
    const float* gk = (const float*)d_in[7];
    const float* bk = (const float*)d_in[8];
    const float* Wv = (const float*)d_in[9];
    const float* gv = (const float*)d_in[10];
    const float* bv = (const float*)d_in[11];
    const float* Wo = (const float*)d_in[12];
    const float* go = (const float*)d_in[13];
    const float* bo = (const float*)d_in[14];
    float* out = (float*)d_out;

    // bf16 workspace: qh | kh | vh | o  (each 64*12*8192 = 6.29M elements)
    const size_t seg = (size_t)BHv * Sv * Dv;             // elements
    unsigned short* qh  = (unsigned short*)d_ws;
    unsigned short* kh  = qh + seg;
    unsigned short* vh  = kh + seg;
    unsigned short* ows = vh + seg;

    const int projLds = 184832;   // 16K W + 34K x-stage + 128K y + red + g/b
    const int outLds  = 300032;   // 16K W + 18K o-stage + 256K y + red + g/b
    (void)hipFuncSetAttribute((const void*)proj_kernel,
                              hipFuncAttributeMaxDynamicSharedMemorySize, projLds);
    (void)hipFuncSetAttribute((const void*)outproj_kernel,
                              hipFuncAttributeMaxDynamicSharedMemorySize, outLds);

    proj_kernel<<<dim3(BSv, 3), 256, projLds, stream>>>(
        q, k, v, Wq, gq, bq, Wk, gk, bk, Wv, gv, bv, qh, kh, vh);
    attn_kernel<<<dim3(BHv), 256, 0, stream>>>(qh, kh, vh, ows);
    outproj_kernel<<<dim3(BSv), 256, outLds, stream>>>(ows, Wo, go, bo, out);
}
